// Network_83648783057298
// MI455X (gfx1250) — matrix operand order repaired and checked by reading, not compiled, not run on hardware
//
#include <hip/hip_runtime.h>

typedef __attribute__((ext_vector_type(16))) _Float16 v16h;
typedef __attribute__((ext_vector_type(8)))  float    v8f;

// ---- Loihi neuron constants ----
#define THETA_F   5120.0f     // 80 * 64
#define I_DEC_F   0.75f       // (4096-1024)/4096
#define V_DEC_F   0.96875f    // (4096-128)/4096
#define W_SCALE_F 64.0f
#define POOL_W_F  88.0f       // 1.1 * 80

__device__ __forceinline__ float lif_step(float x, float& u, float& v, float& ref) {
  u = I_DEC_F * u + W_SCALE_F * x;
  v = (ref > 0.f) ? 0.f : (V_DEC_F * v + u);
  float s = (v >= THETA_F) ? 1.f : 0.f;
  v *= (1.f - s);
  ref = (s > 0.f) ? 1.f : fmaxf(ref - 1.f, 0.f);
  return s;
}

// -------------------------------------------------------------------------
// Implicit-GEMM conv with LDS-staged activations (plain GEMM when KH=KW=1).
// sp  : activations [T,B,H,W,Cin] f16
// wp  : packed weights [OCp][Kp] f16, row-major K contiguous, zero padded
// wsum: output [T,B,OH,OW,OCst] f32
// One 256-thread workgroup stages G images into LDS (vectorized B128 copy),
// then its 8 waves sweep all 16x16 WMMA tiles of the group
// (A gathered from LDS, B streamed from global with prefetch).
// -------------------------------------------------------------------------
template<int Cin,int H,int W,int OCst,int OCp,int KH,int KW,int PAD,
         int OH,int OW,int Kp,int G>
__global__ __launch_bounds__(256) void conv_wmma(
    const _Float16* __restrict__ sp, const _Float16* __restrict__ wp,
    float* __restrict__ wsum)
{
  constexpr int Kreal = Cin * KH * KW;
  constexpr int IMG   = H * W * Cin;        // halves per image
  constexpr int Mg    = G * OH * OW;        // rows per group
  constexpr int MT    = Mg / 16;            // m-tiles per group
  constexpr int NT    = MT * (OCp / 16);    // total tiles per group
  static_assert(Mg % 16 == 0, "group rows must be multiple of 16");
  static_assert((G * IMG) % 8 == 0, "stage size must be multiple of 8 halves");

  __shared__ __align__(16) _Float16 smem[G * IMG];

  const long grp = blockIdx.x;              // images [grp*G, grp*G+G)

  // ---- cooperative coalesced stage: global -> LDS, 16B per thread-iter ----
  {
    const uint4* __restrict__ gsrc = (const uint4*)(sp + grp * (long)(G * IMG));
    uint4* __restrict__ ldst = (uint4*)smem;
    for (int i = threadIdx.x; i < (G * IMG) / 8; i += 256) ldst[i] = gsrc[i];
  }
  __syncthreads();

  const int wave  = threadIdx.x >> 5;
  const int lane  = threadIdx.x & 31;
  const int khalf = lane >> 4;              // 0 or 1
  const int l16   = lane & 15;

  for (int tt = wave; tt < NT; tt += 8) {   // wave-uniform tile loop
    const int mt = tt % MT;
    const int nt = tt / MT;

    // per-lane A row (local)
    const int mloc = mt * 16 + l16;
    const int img  = mloc / (OH * OW);
    const int rm   = mloc % (OH * OW);
    const int oy   = rm / OW;
    const int ox   = rm % OW;
    const _Float16* __restrict__ srow = smem + img * IMG;

    // per-lane B column; prefetch the row before the K loop
    const int n = nt * 16 + l16;
    const _Float16* __restrict__ wrow = wp + (long)n * Kp;
    __builtin_prefetch(wrow, 0, 0);

    v8f acc = {};
    for (int kb = 0; kb < Kp; kb += 32) {
      // A fragment (contiguous k order, same as B): lanes 0-15 K{kb..+15},
      // lanes 16-31 K{kb+16..+31} -- gathered from LDS
      v16h afrag;
#pragma unroll
      for (int j = 0; j < 16; ++j) {
        const int k = kb + khalf * 16 + j;
        _Float16 va = (_Float16)0.0f;
        if (k < Kreal) {
          const int kw = k % KW;
          const int t2 = k / KW;
          const int kh = t2 % KH;
          const int ci = t2 / KH;
          const int iy = oy + kh - PAD;
          const int ix = ox + kw - PAD;
          if (iy >= 0 && iy < H && ix >= 0 && ix < W)
            va = srow[(iy * W + ix) * Cin + ci];
        }
        afrag[j] = va;
      }
      // B fragment (k order matching A): contiguous 16 halves, 32B aligned
      const v16h bfrag = *(const v16h*)(wrow + kb + khalf * 16);

      acc = __builtin_amdgcn_wmma_f32_16x16x32_f16(false, afrag, false, bfrag,
                                                   (short)0, acc, false, false);
    }

    // D write (C/D layout): VGPR r -> M = r + 8*khalf, N = l16
    if (n < OCst) {
#pragma unroll
      for (int r = 0; r < 8; ++r) {
        const long mm = grp * Mg + mt * 16 + khalf * 8 + r;
        wsum[mm * OCst + n] = acc[r];
      }
    }
  }
}

// -------------------------------------------------------------------------
// LIF scan over T (sequential) with 1-step delay shift; one thread = neuron
// -------------------------------------------------------------------------
__global__ void lif_scan(const float* __restrict__ wsum, _Float16* __restrict__ spk,
                         int stride, int T) {
  int j = blockIdx.x * blockDim.x + threadIdx.x;
  if (j >= stride) return;
  float u = 0.f, v = 0.f, ref = 0.f, prev = 0.f;
  for (int t = 0; t < T; ++t) {
    spk[(size_t)t * stride + j] = (_Float16)prev;              // delay_shift
    prev = lif_step(wsum[(size_t)t * stride + j], u, v, ref);
  }
}

// fused 2x2 sum-pool (weight POOL_W) + LIF scan + delay shift
__global__ void pool_lif(const _Float16* __restrict__ spk_in,
                         _Float16* __restrict__ spk_out,
                         int Bsz, int OH, int OW, int C, int T) {
  const int PH = OH / 2, PW = OW / 2;
  const int total = Bsz * PH * PW * C;
  int j = blockIdx.x * blockDim.x + threadIdx.x;
  if (j >= total) return;
  int c  = j % C;  int t1 = j / C;
  int px = t1 % PW; t1 /= PW;
  int py = t1 % PH; int b = t1 / PH;
  const size_t tstr = (size_t)Bsz * OH * OW * C;
  const size_t row  = (size_t)OW * C;
  float u = 0.f, v = 0.f, ref = 0.f, prev = 0.f;
  for (int t = 0; t < T; ++t) {
    size_t base = (size_t)t * tstr + (((size_t)b * OH + 2 * py) * OW + 2 * px) * C + c;
    float sum = (float)spk_in[base] + (float)spk_in[base + C]
              + (float)spk_in[base + row] + (float)spk_in[base + row + C];
    spk_out[(size_t)t * total + j] = (_Float16)prev;
    prev = lif_step(POOL_W_F * sum, u, v, ref);
  }
}

// final LIF scan writing fp32 output [B, O, T]
__global__ void lif_scan_out(const float* __restrict__ wsum, float* __restrict__ out,
                             int Bsz, int O, int T) {
  const int total = Bsz * O;
  int j = blockIdx.x * blockDim.x + threadIdx.x;
  if (j >= total) return;
  int o = j % O, b = j / O;
  float u = 0.f, v = 0.f, ref = 0.f, prev = 0.f;
  for (int t = 0; t < T; ++t) {
    out[((size_t)b * O + o) * T + t] = prev;
    prev = lif_step(wsum[(size_t)t * total + j], u, v, ref);
  }
}

// -------------------------------------------------------------------------
// weight packing: OIHW fp32 -> [OCp][Kp] f16, zero-padded (K = Cin*KH*KW)
// -------------------------------------------------------------------------
__global__ void pack_w(const float* __restrict__ w, _Float16* __restrict__ wp,
                       int OC, int K, int OCp, int Kp) {
  int i = blockIdx.x * blockDim.x + threadIdx.x;
  if (i >= OCp * Kp) return;
  int nIdx = i / Kp, k = i % Kp;
  float v = (nIdx < OC && k < K) ? w[nIdx * K + k] : 0.f;
  wp[i] = (_Float16)v;
}

// fc1 weights [512, 288(c*9+h*3+w)] -> columns permuted to my (h*3+w)*32+c order
__global__ void pack_w_fc1(const float* __restrict__ w, _Float16* __restrict__ wp) {
  int i = blockIdx.x * blockDim.x + threadIdx.x;
  if (i >= 512 * 288) return;
  int nIdx = i / 288, kp = i % 288;
  int c = kp % 32, s = kp / 32;
  int h = s / 3,  ww = s % 3;
  wp[i] = (_Float16)w[nIdx * 288 + c * 9 + h * 3 + ww];
}

// input [48,4,26,26,128] f32 -> [128,48,26,26,4] f16
__global__ void cvt_input(const float* __restrict__ x, _Float16* __restrict__ sp) {
  long i = (long)blockIdx.x * blockDim.x + threadIdx.x;
  const long total = 128L * 48 * 26 * 26 * 4;
  if (i >= total) return;
  int c = (int)(i % 4);  long t1 = i / 4;
  int w = (int)(t1 % 26); t1 /= 26;
  int h = (int)(t1 % 26); t1 /= 26;
  int b = (int)(t1 % 48); int t = (int)(t1 / 48);
  sp[i] = (_Float16)x[((((long)b * 4 + c) * 26 + h) * 26 + w) * 128 + t];
}

static inline int cdiv(long a, int b) { return (int)((a + b - 1) / b); }

extern "C" void kernel_launch(void* const* d_in, const int* in_sizes, int n_in,
                              void* d_out, int out_size, void* d_ws, size_t ws_size,
                              hipStream_t stream) {
  (void)in_sizes; (void)n_in; (void)out_size; (void)ws_size;
  const float* x_in = (const float*)d_in[0];   // [48,4,26,26,128]
  const float* w1   = (const float*)d_in[1];   // [8,4,5,5]
  const float* w2   = (const float*)d_in[2];   // [16,8,3,3]
  const float* w3   = (const float*)d_in[3];   // [32,16,3,3]
  const float* w4   = (const float*)d_in[4];   // [512,288]
  const float* w5   = (const float*)d_in[5];   // [2,512]
  float* out = (float*)d_out;                  // [48,2,128]
  char*  ws  = (char*)d_ws;

  constexpr int T = 128, B = 48;
  // workspace layout
  constexpr size_t WSUM_ELEMS = 28311552ull;           // conv1 wsum (largest)
  constexpr size_t A_ELEMS    = 16613376ull;           // input f16 (largest in A)
  constexpr size_t B_ELEMS    = 28311552ull;           // conv1 spikes (largest in B)
  float*    wsum = (float*)ws;
  _Float16* bufA = (_Float16*)(ws + WSUM_ELEMS * 4);
  _Float16* bufB = (_Float16*)(ws + WSUM_ELEMS * 4 + A_ELEMS * 2);
  _Float16* wp1  = (_Float16*)(ws + WSUM_ELEMS * 4 + A_ELEMS * 2 + B_ELEMS * 2);
  _Float16* wp2  = wp1 + 16 * 128;
  _Float16* wp3  = wp2 + 16 * 96;
  _Float16* wp4  = wp3 + 32 * 160;
  _Float16* wp5  = wp4 + 512 * 288;

  // ---- pack weights ----
  pack_w<<<cdiv(16 * 128, 256), 256, 0, stream>>>(w1, wp1, 8, 100, 16, 128);
  pack_w<<<cdiv(16 *  96, 256), 256, 0, stream>>>(w2, wp2, 16, 72, 16, 96);
  pack_w<<<cdiv(32 * 160, 256), 256, 0, stream>>>(w3, wp3, 32, 144, 32, 160);
  pack_w_fc1<<<cdiv(512 * 288, 256), 256, 0, stream>>>(w4, wp4);
  pack_w<<<cdiv(16 * 512, 256), 256, 0, stream>>>(w5, wp5, 2, 512, 16, 512);

  // ---- input to f16, time-leading HWC ----
  cvt_input<<<cdiv(128L * 48 * 26 * 26 * 4, 256), 256, 0, stream>>>(x_in, bufA);

  // ---- block 1: conv1 (26->24, pad1) -> LIF -> pool2 (24->12) -> LIF ----
  conv_wmma<4, 26, 26, 8, 16, 5, 5, 1, 24, 24, 128, 1>
      <<<6144, 256, 0, stream>>>(bufA, wp1, wsum);
  lif_scan<<<cdiv(221184, 256), 256, 0, stream>>>(wsum, bufB, 221184, T);
  pool_lif<<<cdiv(55296, 256), 256, 0, stream>>>(bufB, bufA, B, 24, 24, 8, T);

  // ---- block 2: conv2 (12->12, pad1) -> LIF -> pool2 (12->6) -> LIF ----
  conv_wmma<8, 12, 12, 16, 16, 3, 3, 1, 12, 12, 96, 1>
      <<<6144, 256, 0, stream>>>(bufA, wp2, wsum);
  lif_scan<<<cdiv(110592, 256), 256, 0, stream>>>(wsum, bufB, 110592, T);
  pool_lif<<<cdiv(27648, 256), 256, 0, stream>>>(bufB, bufA, B, 12, 12, 16, T);

  // ---- block 3: conv3 (6->6, pad1) -> LIF -> pool2 (6->3) -> LIF ----
  conv_wmma<16, 6, 6, 32, 32, 3, 3, 1, 6, 6, 160, 4>
      <<<1536, 256, 0, stream>>>(bufA, wp3, wsum);
  lif_scan<<<cdiv(55296, 256), 256, 0, stream>>>(wsum, bufB, 55296, T);
  pool_lif<<<cdiv(13824, 256), 256, 0, stream>>>(bufB, bufA, B, 6, 6, 32, T);

  // ---- fc1: [6144,288] x [512,288]^T -> LIF ----
  conv_wmma<288, 1, 1, 512, 512, 1, 1, 0, 1, 1, 288, 16>
      <<<384, 256, 0, stream>>>(bufA, wp4, wsum);
  lif_scan<<<cdiv(24576, 256), 256, 0, stream>>>(wsum, bufB, 24576, T);

  // ---- fc2: [6144,512] x [2,512]^T -> LIF -> output [B,2,T] ----
  conv_wmma<512, 1, 1, 2, 16, 1, 1, 0, 1, 1, 512, 64>
      <<<96, 256, 0, stream>>>(bufB, wp5, wsum);
  lif_scan_out<<<cdiv(96, 256), 256, 0, stream>>>(wsum, out, B, 2, T);
}
